// MixLinear_GEMM_68118181314581
// MI455X (gfx1250) — 16-bit and 8-bit matrix operand order repaired and checked by reading, compiled, not run on hardware
//
#include <hip/hip_runtime.h>
#include <hip/hip_bf16.h>

typedef __attribute__((ext_vector_type(4)))  int    v4i;
typedef __attribute__((ext_vector_type(8)))  int    v8i;
typedef __attribute__((ext_vector_type(8)))  float  v8f;
typedef __attribute__((ext_vector_type(16))) __bf16 v16bf;

static constexpr int M  = 8192;
static constexpr int K  = 4096;
static constexpr int N  = 4096;
static constexpr int FP = 256;
#define QMAX 7.0f

#if defined(__has_builtin)
#if __has_builtin(__builtin_amdgcn_global_load_async_to_lds_b128) && \
    __has_builtin(__builtin_amdgcn_s_wait_asynccnt)
#define HAVE_ASYNC_LDS 1
#endif
#endif

__device__ __forceinline__ void cp16(void* l, const void* g) {
#ifdef HAVE_ASYNC_LDS
  __builtin_amdgcn_global_load_async_to_lds_b128(
      (__attribute__((address_space(1))) v4i*)g,
      (__attribute__((address_space(3))) v4i*)l, 0, 0);
#else
  *(int4*)l = *(const int4*)g;
#endif
}
__device__ __forceinline__ void wait_async_copies() {
#ifdef HAVE_ASYNC_LDS
  __builtin_amdgcn_s_wait_asynccnt(0);
#endif
}

__global__ __launch_bounds__(256) void quant_kernel(
    const float* __restrict__ x, const int* __restrict__ ind,
    signed char* __restrict__ qx, float* __restrict__ outl,
    float* __restrict__ xscale) {
  __shared__ __align__(16) unsigned char mask[K];
  __shared__ __align__(16) float rowbuf[K];
  __shared__ float red[256];
  const int m = blockIdx.x;
  const int t = threadIdx.x;

  *(int4*)(mask + t * 16) = make_int4(0, 0, 0, 0);
  __syncthreads();
  if (t < FP) mask[ind[t]] = 1;
  __syncthreads();

  const float* xr = x + (size_t)m * K;
  float amax = 0.0f;
#pragma unroll
  for (int i = 0; i < K / 256; ++i) {
    int c = t + i * 256;
    float v = xr[c];
    rowbuf[c] = v;
    if (!mask[c]) amax = fmaxf(amax, fabsf(v));
  }
  red[t] = amax;
  __syncthreads();
  for (int s = 128; s > 0; s >>= 1) {
    if (t < s) red[t] = fmaxf(red[t], red[t + s]);
    __syncthreads();
  }
  const float scale = red[0] * (1.0f / QMAX);
  const float inv = 1.0f / scale;
  signed char* qr = qx + (size_t)m * K;
#pragma unroll
  for (int i = 0; i < K / 256; ++i) {
    int c = t + i * 256;
    float v = mask[c] ? 0.0f : rowbuf[c];
    float q = fminf(fmaxf(rintf(v * inv), -8.0f), 7.0f);
    qr[c] = (signed char)(int)q;
  }
  if (t < FP) outl[(size_t)m * FP + t] = rowbuf[ind[t]];
  if (t == 0) xscale[m] = scale;
}

__global__ __launch_bounds__(256) void unpack_kernel(
    const int* __restrict__ qw, signed char* __restrict__ Bp) {
  __shared__ __align__(16) signed char Ls[64][64];
  const int nb = blockIdx.x * 64;
  const int kb = blockIdx.y * 64;
  const int t = threadIdx.x;
#pragma unroll
  for (int i = 0; i < 8; ++i) {
    int idx = t + i * 256;
    int n = idx >> 5;
    int wi = idx & 31;
    int q = qw[(size_t)(nb + n) * (K / 2) + (kb >> 1) + wi];
    int lo = q & 15, hi = (q >> 4) & 15;
    Ls[2 * wi][n]     = (signed char)((lo ^ 8) - 8);
    Ls[2 * wi + 1][n] = (signed char)((hi ^ 8) - 8);
  }
  __syncthreads();
  const int krow = t >> 2, qd = t & 3;
  *(int4*)(Bp + (size_t)(kb + krow) * N + nb + qd * 16) =
      *(const int4*)&Ls[krow][qd * 16];
}

__global__ __launch_bounds__(256) void gemm_kernel(
    const signed char* __restrict__ qx, const signed char* __restrict__ Bp,
    const float* __restrict__ scale_col, const float* __restrict__ wcache,
    const float* __restrict__ outl, const float* __restrict__ xscale,
    float* __restrict__ out) {
  __shared__ __align__(16) signed char As[2][128 * 64];
  __shared__ __align__(16) signed char Bs[2][64 * 128];
  __shared__ __align__(16) __bf16 Abf[128 * 32];
  __shared__ __align__(16) __bf16 Bbf[32 * 128];

  const int t = threadIdx.x;
  const int lane = t & 31;
  const int wave = t >> 5;
  const int wm = wave & 3;
  const int wn = wave >> 2;
  const int m0 = blockIdx.x * 128;
  const int n0 = blockIdx.y * 128;
  const int hl = (lane >> 4) & 1;
  const int ln = lane & 15;

  const int a_row = t >> 1, a_half = t & 1;
  const int b_row = t >> 2, b_q = t & 3;
  const signed char* gA0 = qx + (size_t)(m0 + a_row) * K + a_half * 32;
  const signed char* gB0 = Bp + (size_t)b_row * N + n0 + b_q * 32;

  const v8i zacc = {0, 0, 0, 0, 0, 0, 0, 0};
  v8i acc[2][4];
#pragma unroll
  for (int mi = 0; mi < 2; ++mi)
#pragma unroll
    for (int ni = 0; ni < 4; ++ni) acc[mi][ni] = zacc;

  {
    signed char* l = &As[0][a_row * 64 + a_half * 32];
    cp16(l, gA0); cp16(l + 16, gA0 + 16);
    signed char* lb = &Bs[0][b_row * 128 + b_q * 32];
    cp16(lb, gB0); cp16(lb + 16, gB0 + 16);
  }
  wait_async_copies();
  __syncthreads();

#pragma unroll 1
  for (int step = 0; step < K / 64; ++step) {
    const int cur = step & 1;
    if (step + 1 < K / 64) {
      const int kk = (step + 1) * 64;
      const signed char* g = gA0 + kk;
      signed char* l = &As[cur ^ 1][a_row * 64 + a_half * 32];
      cp16(l, g); cp16(l + 16, g + 16);
      const signed char* gb = gB0 + (size_t)kk * N;
      signed char* lb = &Bs[cur ^ 1][b_row * 128 + b_q * 32];
      cp16(lb, gb); cp16(lb + 16, gb + 16);
    }
    if (step + 2 < K / 64) {
      const int kp = (step + 2) * 64;
      __builtin_prefetch(gA0 + kp, 0, 1);
      __builtin_prefetch(gB0 + (size_t)kp * N, 0, 1);
    }
    v8i a[2], b[4];
#pragma unroll
    for (int mi = 0; mi < 2; ++mi) {
      int row = wm * 32 + mi * 16 + ln;
      const signed char* p = &As[cur][row * 64 + hl * 8];
      union { v8i v; unsigned long long q[4]; } u;
      u.q[0] = *(const unsigned long long*)(p);
      u.q[1] = *(const unsigned long long*)(p + 16);
      u.q[2] = *(const unsigned long long*)(p + 32);
      u.q[3] = *(const unsigned long long*)(p + 48);
      a[mi] = u.v;
    }
#pragma unroll
    for (int ni = 0; ni < 4; ++ni) {
      int nb = wn * 64 + ni * 16;
      union { v8i v; signed char c[32]; } u;
#pragma unroll
      for (int s = 0; s < 32; ++s)
        u.c[s] = Bs[cur][((s >> 3) * 16 + hl * 8 + (s & 7)) * 128 + nb + ln];
      b[ni] = u.v;
    }
#pragma unroll
    for (int mi = 0; mi < 2; ++mi)
#pragma unroll
      for (int ni = 0; ni < 4; ++ni)
        acc[mi][ni] = __builtin_amdgcn_wmma_i32_16x16x64_iu8(
            true, a[mi], true, b[ni], acc[mi][ni], false, false);
    wait_async_copies();
    __syncthreads();
  }

  v8f cf[2][4];
  float sc[4];
#pragma unroll
  for (int ni = 0; ni < 4; ++ni) sc[ni] = scale_col[n0 + wn * 64 + ni * 16 + ln];
#pragma unroll
  for (int mi = 0; mi < 2; ++mi) {
    float xs[8];
#pragma unroll
    for (int r = 0; r < 8; ++r)
      xs[r] = xscale[m0 + wm * 32 + mi * 16 + hl * 8 + r];
#pragma unroll
    for (int ni = 0; ni < 4; ++ni)
#pragma unroll
      for (int r = 0; r < 8; ++r)
        cf[mi][ni][r] = (float)acc[mi][ni][r] * xs[r] * sc[ni];
  }

#pragma unroll 1
  for (int kk = 0; kk < FP; kk += 32) {
#pragma unroll
    for (int i = 0; i < 16; ++i) {
      int idx = t + i * 256;
      int mr = idx >> 5, f = idx & 31;
      Abf[mr * 32 + f] = (__bf16)outl[(size_t)(m0 + mr) * FP + kk + f];
    }
#pragma unroll
    for (int i = 0; i < 16; ++i) {
      int idx = t + i * 256;
      int f = idx >> 7, n = idx & 127;
      Bbf[n * 32 + f] = (__bf16)wcache[(size_t)(n0 + n) * FP + kk + f];
    }
    __syncthreads();

    v16bf ab[2], bb[4];
#pragma unroll
    for (int mi = 0; mi < 2; ++mi) {
      int row = wm * 32 + mi * 16 + ln;
      const __bf16* p = Abf + row * 32 + hl * 8;
      union { v16bf v; int4 h[2]; } u;
      u.h[0] = *(const int4*)(p);
      u.h[1] = *(const int4*)(p + 16);
      ab[mi] = u.v;
    }
#pragma unroll
    for (int ni = 0; ni < 4; ++ni) {
      int nb = wn * 64 + ni * 16;
      const __bf16* p = Bbf + (nb + ln) * 32 + hl * 8;
      union { v16bf v; int4 h[2]; } u;
      u.h[0] = *(const int4*)(p);
      u.h[1] = *(const int4*)(p + 16);
      bb[ni] = u.v;
    }
#pragma unroll
    for (int mi = 0; mi < 2; ++mi)
#pragma unroll
      for (int ni = 0; ni < 4; ++ni)
        cf[mi][ni] = __builtin_amdgcn_wmma_f32_16x16x32_bf16(
            false, ab[mi], false, bb[ni], (short)0, cf[mi][ni], false, false);
    __syncthreads();
  }

#pragma unroll
  for (int mi = 0; mi < 2; ++mi)
#pragma unroll
    for (int ni = 0; ni < 4; ++ni) {
      int col = n0 + wn * 64 + ni * 16 + ln;
#pragma unroll
      for (int r = 0; r < 8; ++r) {
        int row = m0 + wm * 32 + mi * 16 + hl * 8 + r;
        out[(size_t)row * N + col] = cf[mi][ni][r];
      }
    }
}

extern "C" void kernel_launch(void* const* d_in, const int* in_sizes, int n_in,
                              void* d_out, int out_size, void* d_ws, size_t ws_size,
                              hipStream_t stream) {
  const float* x         = (const float*)d_in[0];
  const int*   qw        = (const int*)d_in[1];
  const float* scale_col = (const float*)d_in[2];
  const float* wcache    = (const float*)d_in[3];
  const int*   ind       = (const int*)d_in[4];
  float* out = (float*)d_out;

  char* ws = (char*)d_ws;
  const size_t OFF_QX   = 0;
  const size_t OFF_OUTL = (size_t)M * K;
  const size_t OFF_XS   = OFF_OUTL + (size_t)M * FP * 4;
  const size_t OFF_BP   = OFF_XS + 65536;
  signed char* qx   = (signed char*)(ws + OFF_QX);
  float* outl       = (float*)(ws + OFF_OUTL);
  float* xscl       = (float*)(ws + OFF_XS);
  signed char* Bpw  = (signed char*)(ws + OFF_BP);

  quant_kernel<<<M, 256, 0, stream>>>(x, ind, qx, outl, xscl);
  dim3 ugrid(N / 64, K / 64);
  unpack_kernel<<<ugrid, 256, 0, stream>>>(qw, Bpw);
  dim3 grid(M / 128, N / 128);
  gemm_kernel<<<grid, 256, 0, stream>>>(qx, Bpw, scale_col, wcache, outl, xscl, out);
}
